// GlobalGraphAttention_91250875171738
// MI455X (gfx1250) — matrix operand order repaired and checked by reading, not compiled, not run on hardware
//
#include <hip/hip_runtime.h>

// ---------------------------------------------------------------------------
// GlobalGraphAttention forward for MI455X (gfx1250, wave32, WMMA)
// B=8, N=1024, D=512, H=8, HD=64. All matmuls on v_wmma_f32_16x16x32_f16.
// v2: 16x32 GEMM tiles (A-frag reuse), 32-query attention waves (K/V frag
//     reuse), non-temporal loads for the 256MB bias stream + one-shot inputs.
// ---------------------------------------------------------------------------

typedef __attribute__((ext_vector_type(16))) _Float16 v16h;
typedef __attribute__((ext_vector_type(8)))  _Float16 v8h;
typedef __attribute__((ext_vector_type(8)))  float    v8f;

#define WMMA_F16(a, b, c) \
  __builtin_amdgcn_wmma_f32_16x16x32_f16(false, (a), false, (b), (short)0, (c), false, false)

// ---- fragment loaders (contiguous k order, same for A and B) ----------
// A 16x32 f16: lane m=l&15, hi=l>>4; halves 0..15 -> K=16*hi..16*hi+15
static __device__ __forceinline__ v16h load_a_frag(const _Float16* __restrict__ base,
                                                   int ld, int row0, int k0, int lane) {
  const int m = lane & 15, hi = lane >> 4;
  const _Float16* p = base + (size_t)(row0 + m) * ld + k0 + 16 * hi;
  return *(const v16h*)p;
}

// B 32x16 where B = W^T and W is row-major [Nout x K]:
// lane n=l&15, hi=l>>4 holds halves K = k0 + 16*hi + (0..15) of column (n0+n).
static __device__ __forceinline__ v16h load_b_frag_wt(const _Float16* __restrict__ W,
                                                      int ld, int n0, int k0, int lane) {
  const int n = lane & 15, hi = lane >> 4;
  return *(const v16h*)(W + (size_t)(n0 + n) * ld + k0 + 16 * hi);
}

// ---- f32 -> f16 conversion (NT loads: sources are read exactly once) ------
__global__ void __launch_bounds__(256) cvt_f32_f16(const float* __restrict__ src,
                                                   _Float16* __restrict__ dst, int n) {
  int i = blockIdx.x * 256 + threadIdx.x;
  if (i < n) dst[i] = (_Float16)__builtin_nontemporal_load(src + i);
}

// ---- kernel 1: qkv = x @ qkv_w^T + qkv_b, scatter to q16/k16/vT16 ---------
// M=8192, Nout=1536, K=512. One wave per 16x32 tile (A-frag reused twice).
__global__ void __launch_bounds__(256) qkv_kernel(const _Float16* __restrict__ x16,
                                                  const _Float16* __restrict__ w16,
                                                  const float* __restrict__ qb,
                                                  _Float16* __restrict__ q16,
                                                  _Float16* __restrict__ k16,
                                                  _Float16* __restrict__ vT16) {
  const int lane = threadIdx.x & 31;
  const int w = blockIdx.x * 8 + (threadIdx.x >> 5);
  const int mt = w / 48, ntp = w % 48;  // 512 x 48 (16-row x 32-col) tiles
  v8f acc[2] = {{}, {}};
#pragma unroll 4
  for (int k0 = 0; k0 < 512; k0 += 32) {
    const v16h a = load_a_frag(x16, 512, mt * 16, k0, lane);
    acc[0] = WMMA_F16(a, load_b_frag_wt(w16, 512, ntp * 32, k0, lane), acc[0]);
    acc[1] = WMMA_F16(a, load_b_frag_wt(w16, 512, ntp * 32 + 16, k0, lane), acc[1]);
  }
  const int n = lane & 15, hi = lane >> 4;
#pragma unroll
  for (int sub = 0; sub < 2; ++sub) {
    const int col = ntp * 32 + sub * 16 + n;
    const float bias = qb[col];
    const int which = col >> 9;  // 0=q 1=k 2=v
    const int rem = col & 511;
    const int h = rem >> 6, hd = rem & 63;
#pragma unroll
    for (int j = 0; j < 8; ++j) {
      const int m = mt * 16 + j + 8 * hi;
      const int bidx = m >> 10, node = m & 1023;
      const float v = acc[sub][j] + bias;
      if (which == 0)
        q16[(((size_t)bidx * 8 + h) * 1024 + node) * 64 + hd] = (_Float16)(v * 0.125f);
      else if (which == 1)
        k16[(((size_t)bidx * 8 + h) * 1024 + node) * 64 + hd] = (_Float16)v;
      else
        vT16[(((size_t)bidx * 8 + h) * 64 + hd) * 1024 + node] = (_Float16)v;
    }
  }
}

// ---- kernel 2: flash attention, 32 queries x all 1024 keys per wave -------
// K/V B-fragments are loaded once per 32-key step and reused by both 16-row
// query tiles. Bias is streamed with non-temporal loads (256MB, read once).
__global__ void __launch_bounds__(256) attn_kernel(const _Float16* __restrict__ q16,
                                                   const _Float16* __restrict__ k16,
                                                   const _Float16* __restrict__ vT16,
                                                   const float* __restrict__ bias,
                                                   const unsigned char* __restrict__ mask,
                                                   _Float16* __restrict__ attn16) {
  __shared__ __align__(32) _Float16 pbuf_all[8 * 32 * 32];  // 16 KB, 2KB/wave
  const int lane = threadIdx.x & 31;
  const int wib = threadIdx.x >> 5;
  const int w = blockIdx.x * 8 + wib;  // 2048 waves total
  const int qt = w & 31;
  const int h = (w >> 5) & 7;
  const int b = w >> 8;
  const int n = lane & 15, hi = lane >> 4;

  const _Float16* qp = q16 + (size_t)(b * 8 + h) * 1024 * 64;
  const _Float16* kp = k16 + (size_t)(b * 8 + h) * 1024 * 64;
  const _Float16* vp = vT16 + (size_t)(b * 8 + h) * 64 * 1024;
  const float* bp0 = bias + ((size_t)(b * 8 + h) * 1024 + qt * 32) * 1024;
  const unsigned char* mb = mask + b * 1024;
  _Float16* pb = pbuf_all + wib * (32 * 32);

  v16h qa[2][2];
#pragma unroll
  for (int rt = 0; rt < 2; ++rt) {
    qa[rt][0] = load_a_frag(qp, 64, qt * 32 + rt * 16, 0, lane);
    qa[rt][1] = load_a_frag(qp, 64, qt * 32 + rt * 16, 32, lane);
  }
  float mrow[2][8], lrow[2][8];
  v8f O[2][4];
#pragma unroll
  for (int rt = 0; rt < 2; ++rt) {
#pragma unroll
    for (int j = 0; j < 8; ++j) { mrow[rt][j] = -1e30f; lrow[rt][j] = 0.0f; }
    const v8f z = {};
#pragma unroll
    for (int t = 0; t < 4; ++t) O[rt][t] = z;
  }

  for (int kt = 0; kt < 1024; kt += 32) {
    // shared K fragments for both query row-tiles
    const v16h bk00 = load_b_frag_wt(kp, 64, kt, 0, lane);
    const v16h bk01 = load_b_frag_wt(kp, 64, kt, 32, lane);
    const v16h bk10 = load_b_frag_wt(kp, 64, kt + 16, 0, lane);
    const v16h bk11 = load_b_frag_wt(kp, 64, kt + 16, 32, lane);
    const int key0 = kt + n, key1 = kt + 16 + n;
    const bool kv0 = mb[key0] != 0;
    const bool kv1 = mb[key1] != 0;

#pragma unroll
    for (int rt = 0; rt < 2; ++rt) {
      v8f s0 = {}, s1 = {};
      s0 = WMMA_F16(qa[rt][0], bk00, s0);
      s0 = WMMA_F16(qa[rt][1], bk01, s0);
      s1 = WMMA_F16(qa[rt][0], bk10, s1);
      s1 = WMMA_F16(qa[rt][1], bk11, s1);

#pragma unroll
      for (int j = 0; j < 8; ++j) {
        const int row = j + 8 * hi;
        const float* brow = bp0 + (size_t)(rt * 16 + row) * 1024;
        float a0 = kv0 ? s0[j] + __builtin_nontemporal_load(brow + key0) : -1e30f;
        float a1 = kv1 ? s1[j] + __builtin_nontemporal_load(brow + key1) : -1e30f;
        // row max across the 16 lanes of this half-wave (xor masks stay in-group)
        float t = fmaxf(a0, a1);
        t = fmaxf(t, __shfl_xor(t, 1, 32));
        t = fmaxf(t, __shfl_xor(t, 2, 32));
        t = fmaxf(t, __shfl_xor(t, 4, 32));
        t = fmaxf(t, __shfl_xor(t, 8, 32));
        const float mnew = fmaxf(mrow[rt][j], t);
        const float alpha = __expf(mrow[rt][j] - mnew);
        const float p0 = __expf(a0 - mnew);
        const float p1 = __expf(a1 - mnew);
        float rs = p0 + p1;
        rs += __shfl_xor(rs, 1, 32);
        rs += __shfl_xor(rs, 2, 32);
        rs += __shfl_xor(rs, 4, 32);
        rs += __shfl_xor(rs, 8, 32);
        mrow[rt][j] = mnew;
        lrow[rt][j] = lrow[rt][j] * alpha + rs;
        O[rt][0][j] *= alpha; O[rt][1][j] *= alpha;
        O[rt][2][j] *= alpha; O[rt][3][j] *= alpha;
        // stash P in C-layout to per-wave LDS tile (row-major 32x32)
        pb[(rt * 16 + row) * 32 + n] = (_Float16)p0;
        pb[(rt * 16 + row) * 32 + 16 + n] = (_Float16)p1;
      }
    }
    // re-read P as A fragments (same-wave DS ops are in order); shared V frags
    const v16h pa0 = load_a_frag(pb, 32, 0, 0, lane);
    const v16h pa1 = load_a_frag(pb, 32, 16, 0, lane);
#pragma unroll
    for (int t = 0; t < 4; ++t) {
      const v16h bv = load_b_frag_wt(vp, 1024, t * 16, kt, lane);
      O[0][t] = WMMA_F16(pa0, bv, O[0][t]);
      O[1][t] = WMMA_F16(pa1, bv, O[1][t]);
    }
  }

#pragma unroll
  for (int rt = 0; rt < 2; ++rt) {
#pragma unroll
    for (int j = 0; j < 8; ++j) {
      const int row = j + 8 * hi;
      const int node = qt * 32 + rt * 16 + row;
      const float inv = (mb[node] != 0) ? (1.0f / lrow[rt][j]) : 0.0f;  // nan_to_num
      _Float16* op = attn16 + ((size_t)b * 1024 + node) * 512 + h * 64 + n;
      op[0]  = (_Float16)(O[rt][0][j] * inv);
      op[16] = (_Float16)(O[rt][1][j] * inv);
      op[32] = (_Float16)(O[rt][2][j] * inv);
      op[48] = (_Float16)(O[rt][3][j] * inv);
    }
  }
}

// ---- kernel 3: out = attn @ out_w^T + out_b (f32 output, NT stores) -------
__global__ void __launch_bounds__(256) out_kernel(const _Float16* __restrict__ a16,
                                                  const _Float16* __restrict__ w16,
                                                  const float* __restrict__ ob,
                                                  float* __restrict__ out) {
  const int lane = threadIdx.x & 31;
  const int w = blockIdx.x * 8 + (threadIdx.x >> 5);
  const int mt = w >> 4, ntp = w & 15;  // 512 x 16 (16-row x 32-col) tiles
  v8f acc[2] = {{}, {}};
#pragma unroll 4
  for (int k0 = 0; k0 < 512; k0 += 32) {
    const v16h a = load_a_frag(a16, 512, mt * 16, k0, lane);
    acc[0] = WMMA_F16(a, load_b_frag_wt(w16, 512, ntp * 32, k0, lane), acc[0]);
    acc[1] = WMMA_F16(a, load_b_frag_wt(w16, 512, ntp * 32 + 16, k0, lane), acc[1]);
  }
  const int n = lane & 15, hi = lane >> 4;
#pragma unroll
  for (int sub = 0; sub < 2; ++sub) {
    const int col = ntp * 32 + sub * 16 + n;
    const float bias = ob[col];
#pragma unroll
    for (int j = 0; j < 8; ++j) {
      float* p = out + (size_t)(mt * 16 + j + 8 * hi) * 512 + col;
      __builtin_nontemporal_store(acc[sub][j] + bias, p);
    }
  }
}

// ---------------------------------------------------------------------------
extern "C" void kernel_launch(void* const* d_in, const int* in_sizes, int n_in,
                              void* d_out, int out_size, void* d_ws, size_t ws_size,
                              hipStream_t stream) {
  const float* x = (const float*)d_in[0];
  const unsigned char* mask = (const unsigned char*)d_in[1];  // jax bool = 1 byte
  const float* attn_bias = (const float*)d_in[2];
  const float* qkv_w = (const float*)d_in[3];
  const float* qkv_b = (const float*)d_in[4];
  const float* out_w = (const float*)d_in[5];
  const float* out_b = (const float*)d_in[6];

  // workspace carve-up (halves); total ~44 MB
  _Float16* ws = (_Float16*)d_ws;
  _Float16* x16 = ws;                        // 4,194,304
  _Float16* wq16 = x16 + 4194304;            //   786,432
  _Float16* wo16 = wq16 + 786432;            //   262,144
  _Float16* q16 = wo16 + 262144;             // 4,194,304
  _Float16* k16 = q16 + 4194304;             // 4,194,304
  _Float16* vT16 = k16 + 4194304;            // 4,194,304
  _Float16* at16 = vT16 + 4194304;           // 4,194,304

  cvt_f32_f16<<<16384, 256, 0, stream>>>(x, x16, 4194304);
  cvt_f32_f16<<<3072, 256, 0, stream>>>(qkv_w, wq16, 786432);
  cvt_f32_f16<<<1024, 256, 0, stream>>>(out_w, wo16, 262144);

  // 512*48 = 24576 tiles, 8 waves/block
  qkv_kernel<<<3072, 256, 0, stream>>>(x16, wq16, qkv_b, q16, k16, vT16);
  // 8*8*32 = 2048 attention waves
  attn_kernel<<<256, 256, 0, stream>>>(q16, k16, vT16, attn_bias, mask, at16);
  // 512*16 = 8192 tiles
  out_kernel<<<1024, 256, 0, stream>>>(at16, wo16, out_b, (float*)d_out);
}
